// GuidedConvStridePENewAfter_9165460209724
// MI455X (gfx1250) — matrix operand order repaired and checked by reading, not compiled, not run on hardware
//
#include <hip/hip_runtime.h>
#include <hip/hip_bf16.h>

typedef __attribute__((ext_vector_type(16))) __bf16 v16bf;
typedef __attribute__((ext_vector_type(8)))  float  v8f;
typedef __attribute__((ext_vector_type(4)))  unsigned int u32x4;
typedef __attribute__((ext_vector_type(8)))  int  i32x8;
typedef __attribute__((ext_vector_type(4)))  int  i32x4;

#define EPS_BN    1e-5f
#define NEG_SLOPE 0.1f

// ---- WMMA fragment index maps (contiguous k order this file uses for both A and B, wave32) ----
// 16-bit A 16x32: lanes 0-15 row M=lane, K=0..15; lanes 16-31 same rows, K=16..31
__device__ __forceinline__ int a_k_of(int lane, int j) {
    return ((lane >> 4) << 4) + j;
}
// 16-bit B 32x16: lanes 0-15 col N=lane, K=0..15; lanes 16-31 col N=lane-16, K=16..31
__device__ __forceinline__ int b_k_of(int lane, int j) {
    return ((lane >> 4) << 4) + j;
}

__device__ __forceinline__ float bn_mean(const float* st, int ch, float invc) {
    return st[2 * ch] * invc;
}
__device__ __forceinline__ float bn_inv(const float* st, int ch, float invc) {
    float m = st[2 * ch] * invc;
    float v = st[2 * ch + 1] * invc - m * m;
    return rsqrtf(v + EPS_BN);
}

// ---- TDM: DMA one weight K-slab W[0:128, ks*32 : ks*32+32] (fp32, row stride CIN) into LDS ----
// D# per CDNA5 ISA ch.8: group0 = {count/lds_addr/global_addr/type}, group1 = {data_size/dims/tiles/strides}
__device__ __forceinline__ void tdm_load_wslab(const float* W, int CIN, int ks, unsigned lds_off) {
#if __has_builtin(__builtin_amdgcn_tensor_load_to_lds)
    unsigned long long gaddr = (unsigned long long)(const void*)(W + (long)ks * 32);
    u32x4 g0;
    g0[0] = 1u;                                                    // count=1, user desc, no gather
    g0[1] = lds_off;                                               // lds_addr (bytes)
    g0[2] = (unsigned)(gaddr & 0xffffffffull);                     // global_addr[31:0]
    g0[3] = (unsigned)((gaddr >> 32) & 0x1ffffffull) | (2u << 30); // global_addr[56:32], type=2 (image)
    i32x8 g1;
    g1[0] = (int)(2u << 16);                                // data_size=4B, workgroup_mask=0
    g1[1] = (int)(((unsigned)CIN & 0xffffu) << 16);         // tensor_dim0[15:0] @bits[63:48]
    g1[2] = (int)((128u << 16) | ((unsigned)CIN >> 16));    // tensor_dim1=128, tensor_dim0[31:16]
    g1[3] = (int)(32u << 16);                               // tile_dim0=32 @bits[127:112]
    g1[4] = 128;                                            // tile_dim1=128
    g1[5] = CIN;                                            // tensor_dim0_stride[31:0]
    g1[6] = 0;                                              // stride hi, tensor_dim1_stride lo
    g1[7] = 0;
    i32x4 gz = {0, 0, 0, 0};
#if __clang_major__ >= 23
    i32x8 gz8 = {0, 0, 0, 0, 0, 0, 0, 0};
    __builtin_amdgcn_tensor_load_to_lds(g0, g1, gz, gz, gz8, 0);
#else
    __builtin_amdgcn_tensor_load_to_lds(g0, g1, gz, gz, 0);
#endif
    __builtin_amdgcn_s_wait_tensorcnt(0);
#endif
}

// ---------------- zero stats ----------------
__global__ void zero_kernel(float* p, int n) {
    int t = blockIdx.x * blockDim.x + threadIdx.x;
    if (t < n) p[t] = 0.f;
}

// ---------------- geometry / VI transform ----------------
__global__ void geom_kernel(const float* __restrict__ dxyz, const float* __restrict__ sxyz,
                            const float* __restrict__ dnorm, const float* __restrict__ snorm,
                            const int* __restrict__ nei, float* __restrict__ viout,
                            int B, int N, int M) {
    long tid = (long)blockIdx.x * blockDim.x + threadIdx.x;
    long total = (long)B * M * 16;
    if (tid >= total) return;
    long bm = tid >> 4;
    long b = bm / M;
    int idx = nei[tid];
    const float* gx = &dxyz[((long)b * N + idx) * 3];
    const float* gn = &dnorm[((long)b * N + idx) * 3];
    const float* sp = &sxyz[bm * 3];
    const float* sn = &snorm[bm * 3];
    float lx = gx[0] - sp[0], ly = gx[1] - sp[1], lz = gx[2] - sp[2];
    float nx = gn[0], ny = gn[1], nz = gn[2];
    float sx = sn[0], sy = sn[1], sz = sn[2];
    float t9 = sqrtf(lx * lx + ly * ly + lz * lz);
    float rin = 1.f / fmaxf(t9, 1e-12f);
    float rx = lx * rin, ry = ly * rin, rz = lz * rin;
    float proj = sx * rx + sy * ry + sz * rz;
    float vx = sx - proj * rx, vy = sy - proj * ry, vz = sz - proj * rz;
    float vn = sqrtf(vx * vx + vy * vy + vz * vz);
    float vq = 1.f / fmaxf(vn, 1e-12f);
    vx *= vq; vy *= vq; vz *= vq;
    float wx = ry * vz - rz * vy, wy = rz * vx - rx * vz, wz = rx * vy - ry * vx;
    float wn = sqrtf(wx * wx + wy * wy + wz * wz);
    float wq = 1.f / fmaxf(wn, 1e-12f);
    wx *= wq; wy *= wq; wz *= wq;
    float t1 = nx * sx + ny * sy + nz * sz;
    float t3 = rx * nx + ry * ny + rz * nz;
    float t4 = lx * sx + ly * sy + lz * sz;
    float t6 = nx * vx + ny * vy + nz * vz;
    float t7 = nx * wx + ny * wy + nz * wz;
    float cx = ny * sz - nz * sy, cy = nz * sx - nx * sz, cz = nx * sy - ny * sx;
    float t8 = lx * cx + ly * cy + lz * cz;
    float* o = &viout[tid * 12];
    o[0] = t1; o[1] = proj; o[2] = t3; o[3] = t4; o[4] = t3; o[5] = t6;
    o[6] = t7; o[7] = t8;  o[8] = t9; o[9] = lx; o[10] = ly; o[11] = lz;
}

// ---------------- generic WMMA linear (+optional fused relu(bn()) on input, + BN stat accumulation) ----------------
// IN_MODE: 0 = raw input, 1 = relu((x-mean)*inv) using producer stats
template <int CIN, int COUT, int IN_MODE>
__global__ void lin_wmma(const float* __restrict__ x, const float* __restrict__ in_st, float in_invc,
                         const float* __restrict__ W, const float* __restrict__ bias,
                         float* __restrict__ y, float* __restrict__ out_st, long R) {
    constexpr int KS = (CIN + 31) / 32;
    constexpr int CG = (COUT + 15) / 16;
    constexpr bool FULLK = (CIN % 32 == 0);
    int lane = threadIdx.x & 31;
    int wv = threadIdx.x >> 5;
    long rowBase = ((long)blockIdx.x * 8 + wv) * 16;
    if (rowBase >= R) return;  // wave-uniform: EXEC stays all-ones for WMMA
    long arow = rowBase + (lane & 15);
    int colL = lane & 15;
    v8f acc[CG];
#pragma unroll
    for (int g = 0; g < CG; ++g) acc[g] = (v8f){};
    for (int ks = 0; ks < KS; ++ks) {
        v16bf a;
        if constexpr (FULLK) {
            // K-groups are contiguous: 2x (float4,float4) -> global_load_b128
            long base = arow * CIN + ks * 32 + ((lane >> 4) << 4);
            float4 f0 = *(const float4*)&x[base];
            float4 f1 = *(const float4*)&x[base + 4];
            float4 f2 = *(const float4*)&x[base + 8];
            float4 f3 = *(const float4*)&x[base + 12];
            float t[16] = {f0.x, f0.y, f0.z, f0.w, f1.x, f1.y, f1.z, f1.w,
                           f2.x, f2.y, f2.z, f2.w, f3.x, f3.y, f3.z, f3.w};
#pragma unroll
            for (int j = 0; j < 16; ++j) {
                float v = t[j];
                if (IN_MODE == 1) {
                    int k = ks * 32 + a_k_of(lane, j);
                    v = fmaxf((v - bn_mean(in_st, k, in_invc)) * bn_inv(in_st, k, in_invc), 0.f);
                }
                a[j] = (__bf16)v;
            }
        } else {
#pragma unroll
            for (int j = 0; j < 16; ++j) {
                int k = ks * 32 + a_k_of(lane, j);
                float v = 0.f;
                if (k < CIN) {
                    v = x[arow * CIN + k];
                    if (IN_MODE == 1)
                        v = fmaxf((v - bn_mean(in_st, k, in_invc)) * bn_inv(in_st, k, in_invc), 0.f);
                }
                a[j] = (__bf16)v;
            }
        }
#pragma unroll
        for (int g = 0; g < CG; ++g) {
            v16bf bf;
            int col = g * 16 + colL;
            if (FULLK && col < COUT) {
                const float* wp = &W[(long)col * CIN + ks * 32 + ((lane >> 4) << 4)];
                float4 b0 = *(const float4*)wp;
                float4 b1 = *(const float4*)(wp + 4);
                float4 b2 = *(const float4*)(wp + 8);
                float4 b3 = *(const float4*)(wp + 12);
                float t[16] = {b0.x, b0.y, b0.z, b0.w, b1.x, b1.y, b1.z, b1.w,
                               b2.x, b2.y, b2.z, b2.w, b3.x, b3.y, b3.z, b3.w};
#pragma unroll
                for (int j = 0; j < 16; ++j) bf[j] = (__bf16)t[j];
            } else {
#pragma unroll
                for (int j = 0; j < 16; ++j) {
                    int k = ks * 32 + b_k_of(lane, j);
                    float v = ((FULLK || k < CIN) && col < COUT) ? W[(long)col * CIN + k] : 0.f;
                    bf[j] = (__bf16)v;
                }
            }
            acc[g] = __builtin_amdgcn_wmma_f32_16x16x32_bf16(
                false, a, false, bf, (short)0, acc[g], false, false);
        }
    }
#pragma unroll
    for (int g = 0; g < CG; ++g) {
        int col = g * 16 + colL;
        if (col < COUT) {
            float bv = bias[col];
            float s = 0.f, s2 = 0.f;
#pragma unroll
            for (int r = 0; r < 8; ++r) {
                int m = (lane < 16) ? r : (r + 8);
                float v = acc[g][r] + bv;
                y[(rowBase + m) * COUT + col] = v;
                s += v;
                s2 += v * v;
            }
            atomicAdd(&out_st[2 * col], s);
            atomicAdd(&out_st[2 * col + 1], s2);
        }
    }
}

// ---------------- WMMA linear with TDM-staged weights (COUT=128, CIN%32==0) ----------------
template <int CIN>
__global__ void lin_wmma_tdm(const float* __restrict__ x,
                             const float* __restrict__ W, const float* __restrict__ bias,
                             float* __restrict__ y, float* __restrict__ out_st, long R) {
    constexpr int KS = CIN / 32;
    constexpr int CG = 8;  // COUT = 128
    __shared__ __align__(16) float wslab[128 * 32];  // one K-slab of weights (16 KB)
    int lane = threadIdx.x & 31;
    int wv = threadIdx.x >> 5;
    long rowBase = ((long)blockIdx.x * 8 + wv) * 16;
    bool active = rowBase < R;  // wave-uniform
    long arow = rowBase + (lane & 15);
    int colL = lane & 15;
    unsigned lds_off = (unsigned)(uintptr_t)(void*)wslab;  // LDS aperture: addr[31:0] = LDS offset
    v8f acc[CG];
#pragma unroll
    for (int g = 0; g < CG; ++g) acc[g] = (v8f){};
    for (int ks = 0; ks < KS; ++ks) {
#if __has_builtin(__builtin_amdgcn_tensor_load_to_lds)
        if (wv == 0) tdm_load_wslab(W, CIN, ks, lds_off);  // issue + s_wait_tensorcnt(0) on wave 0
#else
        for (int i = threadIdx.x; i < 128 * 32; i += 256)
            wslab[i] = W[(long)(i >> 5) * CIN + ks * 32 + (i & 31)];
#endif
        __syncthreads();  // slab visible to all waves
        if (active) {
            long base = arow * CIN + ks * 32 + ((lane >> 4) << 4);
            float4 f0 = *(const float4*)&x[base];
            float4 f1 = *(const float4*)&x[base + 4];
            float4 f2 = *(const float4*)&x[base + 8];
            float4 f3 = *(const float4*)&x[base + 12];
            if (ks + 1 < KS) __builtin_prefetch(&x[base + 32], 0, 3);  // global_prefetch_b8
            v16bf a;
            float ta[16] = {f0.x, f0.y, f0.z, f0.w, f1.x, f1.y, f1.z, f1.w,
                            f2.x, f2.y, f2.z, f2.w, f3.x, f3.y, f3.z, f3.w};
#pragma unroll
            for (int j = 0; j < 16; ++j) a[j] = (__bf16)ta[j];
#pragma unroll
            for (int g = 0; g < CG; ++g) {
                int col = g * 16 + colL;
                const float* wp = &wslab[col * 32 + ((lane >> 4) << 4)];
                float4 b0 = *(const float4*)wp;     // ds_load_b128
                float4 b1 = *(const float4*)(wp + 4);
                float4 b2 = *(const float4*)(wp + 8);
                float4 b3 = *(const float4*)(wp + 12);
                float tb[16] = {b0.x, b0.y, b0.z, b0.w, b1.x, b1.y, b1.z, b1.w,
                                b2.x, b2.y, b2.z, b2.w, b3.x, b3.y, b3.z, b3.w};
                v16bf bf;
#pragma unroll
                for (int j = 0; j < 16; ++j) bf[j] = (__bf16)tb[j];
                acc[g] = __builtin_amdgcn_wmma_f32_16x16x32_bf16(
                    false, a, false, bf, (short)0, acc[g], false, false);
            }
        }
        __syncthreads();  // all waves done with slab before next DMA overwrites it
    }
    if (!active) return;
#pragma unroll
    for (int g = 0; g < CG; ++g) {
        int col = g * 16 + colL;
        float bv = bias[col];
        float s = 0.f, s2 = 0.f;
#pragma unroll
        for (int r = 0; r < 8; ++r) {
            int m = (lane < 16) ? r : (r + 8);
            float v = acc[g][r] + bv;
            y[(rowBase + m) * 128 + col] = v;
            s += v;
            s2 += v * v;
        }
        atomicAdd(&out_st[2 * col], s);
        atomicAdd(&out_st[2 * col + 1], s2);
    }
}

// ---------------- guidance-mix GEMM: sc_raw = ((y_gu - max_k y_gu)*inv_gu) @ gm0_w^T + b ----------------
__global__ void gm0_kernel(const float* __restrict__ gu_raw, const float* __restrict__ gu_st, float gu_invc,
                           const float* __restrict__ W, const float* __restrict__ bias,
                           float* __restrict__ y, float* __restrict__ out_st, long P) {
    __shared__ float lds[8][512];  // per wave: 16 neighbors x 32 ch
    int lane = threadIdx.x & 31;
    int wv = threadIdx.x >> 5;
    long pt = (long)blockIdx.x * 8 + wv;
    bool active = pt < P;
    if (active) {
        int c = lane;  // channel 0..31
        float iv = bn_inv(gu_st, c, gu_invc);
        float vals[16];
        float mx = -3.4e38f;
#pragma unroll
        for (int k = 0; k < 16; ++k) {
            float v = gu_raw[(pt * 16 + k) * 32 + c];
            vals[k] = v;
            mx = fmaxf(mx, v);
        }
#pragma unroll
        for (int k = 0; k < 16; ++k) lds[wv][k * 32 + c] = (vals[k] - mx) * iv;
    }
    __syncthreads();
    if (!active) return;  // wave-uniform
    v16bf a;
#pragma unroll
    for (int j = 0; j < 16; ++j)
        a[j] = (__bf16)lds[wv][(lane & 15) * 32 + a_k_of(lane, j)];
    v16bf bf;
    int col = lane & 15;
#pragma unroll
    for (int j = 0; j < 16; ++j)
        bf[j] = (__bf16)((col < 8) ? W[col * 32 + b_k_of(lane, j)] : 0.f);
    v8f acc = {};
    acc = __builtin_amdgcn_wmma_f32_16x16x32_bf16(false, a, false, bf, (short)0, acc, false, false);
    if (col < 8) {
        float bv = bias[col];
        float s = 0.f, s2 = 0.f;
#pragma unroll
        for (int r = 0; r < 8; ++r) {
            int m = (lane < 16) ? r : (r + 8);
            float v = acc[r] + bv;
            y[(pt * 16 + m) * 8 + col] = v;
            s += v;
            s2 += v * v;
        }
        atomicAdd(&out_st[2 * col], s);
        atomicAdd(&out_st[2 * col + 1], s2);
    }
}

// ---------------- scores: sigmoid(relu(bn(sc_raw)) @ gm1^T + b) ----------------
__global__ void scores_kernel(const float* __restrict__ sc_raw, const float* __restrict__ st, float invc,
                              const float* __restrict__ gw, const float* __restrict__ gb,
                              float* __restrict__ scores, long R) {
    long t = (long)blockIdx.x * blockDim.x + threadIdx.x;
    if (t >= R) return;
    float h[8];
#pragma unroll
    for (int i = 0; i < 8; ++i) {
        float m = bn_mean(st, i, invc), iv = bn_inv(st, i, invc);
        h[i] = fmaxf((sc_raw[t * 8 + i] - m) * iv, 0.f);
    }
#pragma unroll
    for (int o = 0; o < 4; ++o) {
        float s = gb[o];
#pragma unroll
        for (int i = 0; i < 8; ++i) s += h[i] * gw[o * 8 + i];
        scores[t * 4 + o] = 1.f / (1.f + __expf(-s));
    }
}

// ---------------- einsum: nf[c,w] = sum_k gf[k,c]*w[k,w]  (Gf^T @ W via 2 WMMAs/point) ----------------
__global__ void einsum_kernel(const float* __restrict__ fx_raw, const float* __restrict__ u1_st, float u1_invc,
                              const float* __restrict__ w2_raw, const float* __restrict__ wn2_st, float wn_invc,
                              const float* __restrict__ scores, const int* __restrict__ nei,
                              float* __restrict__ nf, int N, int M, long P) {
    __shared__ float gf[8][512];  // 16 k x 32 c
    __shared__ float wm[8][256];  // 16 k x 16 w
    int lane = threadIdx.x & 31;
    int wv = threadIdx.x >> 5;
    long pt = (long)blockIdx.x * 8 + wv;
    bool active = pt < P;
    if (active) {
        int c = lane;  // 0..31
        int head = c >> 3;
        long b = pt / M;
        float m = bn_mean(u1_st, c, u1_invc), iv = bn_inv(u1_st, c, u1_invc);
#pragma unroll
        for (int k = 0; k < 16; ++k) {
            int idx = nei[pt * 16 + k];
            float v = (fx_raw[((long)b * N + idx) * 32 + c] - m) * iv;
            v = (v > 0.f) ? v : NEG_SLOPE * v;
            v *= scores[(pt * 16 + k) * 4 + head];
            gf[wv][k * 32 + c] = v;
        }
        if (lane < 16) {
            int wc = lane;
            float mw = bn_mean(wn2_st, wc, wn_invc), ivw = bn_inv(wn2_st, wc, wn_invc);
#pragma unroll
            for (int k = 0; k < 16; ++k) {
                float v = (w2_raw[(pt * 16 + k) * 16 + wc] - mw) * ivw;
                wm[wv][k * 16 + wc] = fmaxf(v, 0.f);
            }
        }
    }
    __syncthreads();
    if (!active) return;  // wave-uniform
    int wcol = lane & 15;
    v16bf bf;  // B = W (K=16 real + 16 zero-pad)
#pragma unroll
    for (int j = 0; j < 16; ++j)
        bf[j] = (lane < 16) ? (__bf16)wm[wv][j * 16 + wcol] : (__bf16)0.f;
#pragma unroll
    for (int t = 0; t < 2; ++t) {  // A = Gf^T, rows c in [t*16, t*16+16)
        v16bf a;
#pragma unroll
        for (int j = 0; j < 16; ++j) {
            int k = a_k_of(lane, j);
            a[j] = (k < 16) ? (__bf16)gf[wv][k * 32 + (t * 16 + (lane & 15))] : (__bf16)0.f;
        }
        v8f acc = {};
        acc = __builtin_amdgcn_wmma_f32_16x16x32_bf16(false, a, false, bf, (short)0, acc, false, false);
#pragma unroll
        for (int r = 0; r < 8; ++r) {
            int c = t * 16 + ((lane < 16) ? r : (r + 8));
            nf[pt * 512 + (long)c * 16 + wcol] = acc[r];
        }
    }
}

// ---------------- neighborhood max of dense_feats ----------------
__global__ void maxfeat_kernel(const float* __restrict__ df, const int* __restrict__ nei,
                               float* __restrict__ maxf, int N, int M, long total) {
    long t = (long)blockIdx.x * blockDim.x + threadIdx.x;
    if (t >= total) return;
    int c = t & 63;
    long pt = t >> 6;
    long b = pt / M;
    float mx = -3.4e38f;
#pragma unroll 4
    for (int k = 0; k < 16; ++k) {
        int idx = nei[pt * 16 + k];
        mx = fmaxf(mx, df[((long)b * N + idx) * 64 + c]);
    }
    maxf[t] = mx;
}

// ---------------- final: leaky(bn(u2)+bn(us)) ----------------
__global__ void final_kernel(const float* __restrict__ u2_raw, const float* __restrict__ u2_st,
                             const float* __restrict__ us_raw, const float* __restrict__ us_st,
                             float invc, float* __restrict__ out, long total) {
    long t = (long)blockIdx.x * blockDim.x + threadIdx.x;
    if (t >= total) return;
    int c = t & 127;
    float a = (u2_raw[t] - bn_mean(u2_st, c, invc)) * bn_inv(u2_st, c, invc);
    float b = (us_raw[t] - bn_mean(us_st, c, invc)) * bn_inv(us_st, c, invc);
    float v = a + b;
    out[t] = (v > 0.f) ? v : NEG_SLOPE * v;
}

extern "C" void kernel_launch(void* const* d_in, const int* in_sizes, int n_in,
                              void* d_out, int out_size, void* d_ws, size_t ws_size,
                              hipStream_t stream) {
    (void)n_in; (void)out_size; (void)ws_size;
    const float* dxyz = (const float*)d_in[0];
    const float* sxyz = (const float*)d_in[1];
    const float* dfeat = (const float*)d_in[2];
    const float* dnorm = (const float*)d_in[3];
    const float* snorm = (const float*)d_in[4];
    const float* mlp_w = (const float*)d_in[5];
    const float* mlp_b = (const float*)d_in[6];
    const float* gu_w = (const float*)d_in[7];
    const float* gu_b = (const float*)d_in[8];
    const float* gm0_w = (const float*)d_in[9];
    const float* gm0_b = (const float*)d_in[10];
    const float* gm1_w = (const float*)d_in[11];
    const float* gm1_b = (const float*)d_in[12];
    const float* u1_w = (const float*)d_in[13];
    const float* u1_b = (const float*)d_in[14];
    const float* wn0_w = (const float*)d_in[15];
    const float* wn0_b = (const float*)d_in[16];
    const float* wn1_w = (const float*)d_in[17];
    const float* wn1_b = (const float*)d_in[18];
    const float* wn2_w = (const float*)d_in[19];
    const float* wn2_b = (const float*)d_in[20];
    const float* u2_w = (const float*)d_in[21];
    const float* u2_b = (const float*)d_in[22];
    const float* us_w = (const float*)d_in[23];
    const float* us_b = (const float*)d_in[24];
    const int* nei = (const int*)d_in[25];
    float* out = (float*)d_out;

    const int B = 2;
    const int N = in_sizes[0] / (B * 3);
    const int M = in_sizes[1] / (B * 3);
    const long Rn = (long)B * M * 16;  // neighbor samples (800k)
    const long Rp = (long)B * M;       // points (50k)
    const long Rd = (long)B * N;       // dense points (200k)

    float* ws = (float*)d_ws;
    float* vi   = ws;
    float* pe   = vi + 12 * Rn;
    float* gu   = pe + 32 * Rn;
    float* sc   = gu + 32 * Rn;
    float* scor = sc + 8 * Rn;
    float* fx   = scor + 4 * Rn;
    float* u2r  = fx + 32 * Rd;
    float* mxf  = u2r + 128 * Rp;
    float* usr  = mxf + 64 * Rp;
    float* st   = usr + 128 * Rp;
    // buffer reuse (producers fully consumed before overwrite, stream-ordered):
    float* w0 = gu;             // gu free after gm0_kernel
    float* w1 = gu + 8 * Rn;
    float* w2 = gu + 16 * Rn;
    float* nf = pe;             // pe free after gu GEMM; 512*Rp == 32*Rn
    // per-channel {sum, sumsq} slices
    float* sPE = st + 0;    // 32 ch
    float* sGU = st + 64;   // 32 ch
    float* sGM0 = st + 128; // 8 ch
    float* sU1 = st + 144;  // 32 ch
    float* sWN0 = st + 208; // 8 ch
    float* sWN1 = st + 224; // 8 ch
    float* sWN2 = st + 240; // 16 ch
    float* sU2 = st + 272;  // 128 ch
    float* sUS = st + 528;  // 128 ch
    const float iRn = 1.f / (float)Rn;
    const float iRp = 1.f / (float)Rp;
    const float iRd = 1.f / (float)Rd;

    auto tiles = [](long R) { return (int)((R / 16 + 7) / 8); };

    zero_kernel<<<4, 256, 0, stream>>>(st, 784);
    geom_kernel<<<(int)((Rn + 255) / 256), 256, 0, stream>>>(dxyz, sxyz, dnorm, snorm, nei, vi, B, N, M);
    lin_wmma<12, 32, 0><<<tiles(Rn), 256, 0, stream>>>(vi, nullptr, 0.f, mlp_w, mlp_b, pe, sPE, Rn);
    lin_wmma<32, 32, 1><<<tiles(Rn), 256, 0, stream>>>(pe, sPE, iRn, gu_w, gu_b, gu, sGU, Rn);
    gm0_kernel<<<(int)((Rp + 7) / 8), 256, 0, stream>>>(gu, sGU, iRn, gm0_w, gm0_b, sc, sGM0, Rp);
    scores_kernel<<<(int)((Rn + 255) / 256), 256, 0, stream>>>(sc, sGM0, iRn, gm1_w, gm1_b, scor, Rn);
    lin_wmma<64, 32, 0><<<tiles(Rd), 256, 0, stream>>>(dfeat, nullptr, 0.f, u1_w, u1_b, fx, sU1, Rd);
    lin_wmma<12, 8, 0><<<tiles(Rn), 256, 0, stream>>>(vi, nullptr, 0.f, wn0_w, wn0_b, w0, sWN0, Rn);
    lin_wmma<8, 8, 1><<<tiles(Rn), 256, 0, stream>>>(w0, sWN0, iRn, wn1_w, wn1_b, w1, sWN1, Rn);
    lin_wmma<8, 16, 1><<<tiles(Rn), 256, 0, stream>>>(w1, sWN1, iRn, wn2_w, wn2_b, w2, sWN2, Rn);
    einsum_kernel<<<(int)((Rp + 7) / 8), 256, 0, stream>>>(fx, sU1, iRd, w2, sWN2, iRn, scor, nei, nf, N, M, Rp);
    lin_wmma_tdm<512><<<tiles(Rp), 256, 0, stream>>>(nf, u2_w, u2_b, u2r, sU2, Rp);
    maxfeat_kernel<<<(int)((Rp * 64 + 255) / 256), 256, 0, stream>>>(dfeat, nei, mxf, N, M, Rp * 64);
    lin_wmma_tdm<64><<<tiles(Rp), 256, 0, stream>>>(mxf, us_w, us_b, usr, sUS, Rp);
    final_kernel<<<(int)((Rp * 128 + 255) / 256), 256, 0, stream>>>(u2r, sU2, usr, sUS, iRp, out, Rp * 128);
}
